// NeuralTensorInteraction_21620865368735
// MI455X (gfx1250) — matrix operand order repaired and checked by reading, not compiled, not run on hardware
//
#include <hip/hip_runtime.h>
#include <hip/hip_bf16.h>
#include <stdint.h>

#define HID     1024
#define KMAT    16
#define KSTEPS  32      // 1024 / 32 (WMMA K)
#define NCHUNKS 64      // 1024 / 16 (WMMA N)

typedef __attribute__((ext_vector_type(16))) __bf16       v16bf;
typedef __attribute__((ext_vector_type(8)))  float        v8f;
typedef __attribute__((ext_vector_type(4)))  unsigned int v4u;
typedef __attribute__((ext_vector_type(8)))  int          v8i;
typedef __attribute__((ext_vector_type(4)))  int          v4i;

struct Frag32 { uint4 lo, hi; };   // 32 bytes = 16 bf16 = one WMMA A/B fragment per lane

__device__ __forceinline__ unsigned pack2bf(float a, float b) {
  unsigned ua = __builtin_bit_cast(unsigned, a);
  unsigned ub = __builtin_bit_cast(unsigned, b);
  unsigned ra = (ua + 0x7FFFu + ((ua >> 16) & 1u)) >> 16;   // round-to-nearest-even
  unsigned rb = (ub + 0x7FFFu + ((ub >> 16) & 1u)) >> 16;
  return (ra & 0xFFFFu) | (rb << 16);
}

// ---------------------------------------------------------------------------
// Pack A[16384,1024] f32 -> bf16 WMMA A-fragments, K-step interleaved:
// apack[t64][s (K-step)][q (subtile 0..3)][lane 0..31][8 dwords];
// 4 KiB per (t64,s), 128 KiB contiguous per t64.
// A-matrix 16x32 bf16 layout (ISA 7.12.2): lane L holds row m=L%16,
// VGPR j pair K = (j&3)*2 + 8*(L/16) + 16*(j/4).
// ---------------------------------------------------------------------------
__global__ void pack_a_kernel(const float* __restrict__ A, unsigned* __restrict__ apack) {
  const int bx = blockIdx.x;             // 0..32767 = t16*32 + s
  const int t = bx >> 5, s = bx & 31;
  const int t64 = t >> 2, q = t & 3;
  const int L = threadIdx.x;
  const int m = L & 15, h = L >> 4;
  const float* arow = A + (size_t)(t * 16 + m) * HID + s * 32;
  unsigned* dst = apack + ((((size_t)t64 * 32 + s) * 4 + q) * 32 + L) * 8;
#pragma unroll
  for (int j = 0; j < 8; ++j) {
    const int pk = ((j & 3) << 1) + (h << 3) + ((j >> 2) << 4);
    dst[j] = pack2bf(arow[pk], arow[pk + 1]);
  }
}

// ---------------------------------------------------------------------------
// Pack W[16,1024,1024] f32 -> bf16 WMMA B-fragments.
// Layout: wpack[k][c (N-chunk)][s (K-step)][lane][8 dwords], 1 KiB per (k,c,s).
// B-matrix 32x16 bf16 layout: lane L holds col n=L%16,
// VGPR j pair K = (j&3)*2 + 8*(L/16) + 16*(j/4).
// ---------------------------------------------------------------------------
__global__ void pack_w_kernel(const float* __restrict__ W, unsigned* __restrict__ wpack) {
  const int bx = blockIdx.x;             // 0..32767 = k*2048 + c*32 + s
  const int k = bx >> 11;
  const int c = (bx >> 5) & 63;
  const int s = bx & 31;
  const int L = threadIdx.x;
  const int n = (c << 4) + (L & 15);
  const int h = L >> 4;
  const float* wbase = W + (size_t)k * HID * HID + (size_t)(s * 32) * HID + n;
  unsigned* dst = wpack + ((((size_t)k * 64 + c) * 32 + s) * 32 + L) * 8;
#pragma unroll
  for (int j = 0; j < 8; ++j) {
    const int pk = ((j & 3) << 1) + (h << 3) + ((j >> 2) << 4);
    dst[j] = pack2bf(wbase[(size_t)pk * HID], wbase[(size_t)(pk + 1) * HID]);
  }
}

// ---------------------------------------------------------------------------
// Main: block = 64 batch rows (4 subtiles) x one k-group of 8.
// 8 waves; wave w owns k = 8*blockIdx.y + w and all 4 subtiles, so every
// W fragment load feeds 4 WMMAs on 4 independent accumulator chains.
// scores[b,k] = sum_d (A_tile @ W[k])[m,d] * B[b,d], fused per 16-col chunk.
// ---------------------------------------------------------------------------
__global__ __launch_bounds__(256) void ntn_wmma_kernel(
    const unsigned* __restrict__ apack, const unsigned* __restrict__ wpack,
    const float* __restrict__ Bm, float* __restrict__ scores) {
  __shared__ __align__(16) unsigned       sA[32768];    // 128 KiB: packed A, [s][q][L][8]
  __shared__ __align__(16) unsigned short sBt[65536];   // 128 KiB: bf16 B^T [d][m64]
  const int t64 = blockIdx.x;            // 0..255, rows 64*t64 .. 64*t64+63
  const int tid = threadIdx.x;
  const int L   = tid & 31, w = tid >> 5;
  const int h   = L >> 4;
  const int k   = (int)blockIdx.y * 8 + w;

  // ---- Issue TDM copy of the contiguous 128 KiB packed A region ----
#if defined(__gfx1250__) && __has_builtin(__builtin_amdgcn_tensor_load_to_lds) && __has_builtin(__builtin_amdgcn_s_wait_tensorcnt)
  if (w == 0) {
    const unsigned long long ga = (unsigned long long)(uintptr_t)(apack + (size_t)t64 * 32768);
    const unsigned lds = (unsigned)(uintptr_t)&sA[0];   // low 32 bits of generic ptr = LDS byte addr
    v4u g0;
    g0[0] = 1u;                                         // count=1, no gather
    g0[1] = lds;                                        // lds_addr
    g0[2] = (unsigned)ga;                               // global_addr[31:0]
    g0[3] = ((unsigned)(ga >> 32) & 0x01FFFFFFu) | (2u << 30);  // addr[56:32] | type=2
    v8i g1;
    g1[0] = (int)(2u << 16);        // workgroup_mask=0, data_size=4B
    g1[1] = (int)(32768u << 16);    // tensor_dim0 = 32768 dwords (low half of bits 79:48)
    g1[2] = (int)(1u << 16);        // tensor_dim0 hi=0; tensor_dim1 = 1
    g1[3] = (int)(32768u << 16);    // tensor_dim1 hi=0; tile_dim0 = 32768
    g1[4] = 1;                      // tile_dim1 = 1, tile_dim2 = 0
    g1[5] = 32768;                  // tensor_dim0_stride (low 32)
    g1[6] = 0;
    g1[7] = 0;
    const v4i gz4 = {0, 0, 0, 0};
    const v8i gz8 = {0, 0, 0, 0, 0, 0, 0, 0};
    __builtin_amdgcn_tensor_load_to_lds(g0, g1, gz4, gz4, gz8, 0);
  }
#else
  {
    const uint4* src  = (const uint4*)(apack + (size_t)t64 * 32768);
    uint4*       dst4 = (uint4*)sA;
    for (int i = tid; i < 8192; i += 256) dst4[i] = src[i];
  }
#endif

  // ---- Stage B tile transposed as bf16 (overlaps the TDM copy) ----
  {
    const float* brow = Bm + (size_t)t64 * 64 * HID;
    for (int e = tid; e < 65536; e += 256) {
      const int m = e >> 10, d = e & 1023;    // m = 0..63
      const unsigned u = __builtin_bit_cast(unsigned, brow[(size_t)m * HID + d]);
      sBt[(d << 6) | m] = (unsigned short)((u + 0x7FFFu + ((u >> 16) & 1u)) >> 16);
    }
  }
#if defined(__gfx1250__) && __has_builtin(__builtin_amdgcn_tensor_load_to_lds) && __has_builtin(__builtin_amdgcn_s_wait_tensorcnt)
  if (w == 0) __builtin_amdgcn_s_wait_tensorcnt(0);
#endif
  __syncthreads();

  float sacc[4][8];
#pragma unroll
  for (int q = 0; q < 4; ++q)
#pragma unroll
    for (int j = 0; j < 8; ++j) sacc[q][j] = 0.f;

  for (int c = 0; c < NCHUNKS; ++c) {
    v8f acc[4];
#pragma unroll
    for (int q = 0; q < 4; ++q) acc[q] = (v8f){};
    const uint4*    wp = (const uint4*)(wpack + ((((size_t)k * 64 + c) * 32) * 32 + L) * 8);
    const unsigned* as = sA + L * 8;        // K-step base; + s*1024 + q*256 dwords
#pragma unroll 2
    for (int s = 0; s < KSTEPS; ++s) {
      Frag32 fw;
      fw.lo = wp[0]; fw.hi = wp[1];
      wp += 64;                              // 32 lanes * 8 dwords per K-step
      const v16bf bfrag = __builtin_bit_cast(v16bf, fw);
#pragma unroll
      for (int q = 0; q < 4; ++q) {
        Frag32 fa;
        fa.lo = *(const uint4*)(as + q * 256);       // byte offset q*1024
        fa.hi = *(const uint4*)(as + q * 256 + 4);   // byte offset q*1024+16
        const v16bf afrag = __builtin_bit_cast(v16bf, fa);
        acc[q] = __builtin_amdgcn_wmma_f32_16x16x32_bf16(
            false, afrag, false, bfrag, (short)0, acc[q], false, false);
      }
      as += 1024;                            // next K-step block (4 KiB)
    }
    // Fuse the B-dot: C layout -> lane n = L%16, VGPR j -> M = j + 8h.
    const int d = (c << 4) + (L & 15);
#pragma unroll
    for (int q = 0; q < 4; ++q) {
      const uint4 bt = *(const uint4*)(sBt + (d << 6) + (q << 4) + (h << 3));
      const unsigned bu[4] = {bt.x, bt.y, bt.z, bt.w};
#pragma unroll
      for (int j = 0; j < 8; ++j) {
        const unsigned bits = (j & 1) ? (bu[j >> 1] >> 16) : (bu[j >> 1] & 0xFFFFu);
        const float bv = __builtin_bit_cast(float, bits << 16);
        sacc[q][j] += acc[q][j] * bv;
      }
    }
  }

  // Reduce over the 16 lanes of each half (same M set per half) and store.
#pragma unroll
  for (int q = 0; q < 4; ++q) {
#pragma unroll
    for (int j = 0; j < 8; ++j) {
      float v = sacc[q][j];
      v += __shfl_xor(v, 1, 32);
      v += __shfl_xor(v, 2, 32);
      v += __shfl_xor(v, 4, 32);
      v += __shfl_xor(v, 8, 32);
      sacc[q][j] = v;
    }
    if ((L & 15) == 0) {
#pragma unroll
      for (int j = 0; j < 8; ++j)
        scores[((size_t)t64 * 64 + (q << 4) + (h << 3) + j) * KMAT + k] = sacc[q][j];
    }
  }
}

// ---------------------------------------------------------------------------
// Epilogue (memory-bound, full f32): linear term + tanh + output head.
// One wave per batch element.
// ---------------------------------------------------------------------------
__global__ __launch_bounds__(256) void finalize_kernel(
    const float* __restrict__ A, const float* __restrict__ Bm,
    const float* __restrict__ Vw, const float* __restrict__ Vb,
    const float* __restrict__ Ow, const float* __restrict__ Ob,
    const float* __restrict__ scores, float* __restrict__ out) {
  const int tid = threadIdx.x;
  const int L = tid & 31, w = tid >> 5;
  const int b = blockIdx.x * 8 + w;
  const float4* arow = (const float4*)(A + (size_t)b * HID);
  const float4* brow = (const float4*)(Bm + (size_t)b * HID);
  float4 av[8], bv[8];
#pragma unroll
  for (int i = 0; i < 8; ++i) { av[i] = arow[L + 32 * i]; bv[i] = brow[L + 32 * i]; }
  float o = 0.f;
  for (int k = 0; k < KMAT; ++k) {
    const float4* vwa = (const float4*)(Vw + (size_t)k * 2 * HID);
    const float4* vwb = vwa + 256;
    float s = 0.f;
#pragma unroll
    for (int i = 0; i < 8; ++i) {
      const float4 wa = vwa[L + 32 * i], wb = vwb[L + 32 * i];
      s += av[i].x * wa.x + av[i].y * wa.y + av[i].z * wa.z + av[i].w * wa.w;
      s += bv[i].x * wb.x + bv[i].y * wb.y + bv[i].z * wb.z + bv[i].w * wb.w;
    }
    s += __shfl_xor(s, 16, 32);
    s += __shfl_xor(s, 8, 32);
    s += __shfl_xor(s, 4, 32);
    s += __shfl_xor(s, 2, 32);
    s += __shfl_xor(s, 1, 32);
    const float hk = tanhf(scores[(size_t)b * KMAT + k] + s + Vb[k]);
    o += hk * Ow[k];
  }
  if (L == 0) out[b] = o + Ob[0];
}

extern "C" void kernel_launch(void* const* d_in, const int* in_sizes, int n_in,
                              void* d_out, int out_size, void* d_ws, size_t ws_size,
                              hipStream_t stream) {
  (void)in_sizes; (void)n_in; (void)out_size; (void)ws_size;
  const float* A  = (const float*)d_in[0];
  const float* Bm = (const float*)d_in[1];
  const float* W  = (const float*)d_in[2];
  const float* Vw = (const float*)d_in[3];
  const float* Vb = (const float*)d_in[4];
  const float* Ow = (const float*)d_in[5];
  const float* Ob = (const float*)d_in[6];
  float* out = (float*)d_out;

  // Workspace: [scores 1 MiB][Apack 32 MiB][Wpack 32 MiB] = 65 MiB
  char* ws = (char*)d_ws;
  float*    scores = (float*)ws;
  unsigned* apack  = (unsigned*)(ws + (size_t)(1u << 20));
  unsigned* wpack  = (unsigned*)(ws + (size_t)(1u << 20) + ((size_t)32u << 20));

  hipLaunchKernelGGL(pack_a_kernel,   dim3(32768),   dim3(32),  0, stream, A, apack);
  hipLaunchKernelGGL(pack_w_kernel,   dim3(32768),   dim3(32),  0, stream, W, wpack);
  hipLaunchKernelGGL(ntn_wmma_kernel, dim3(256, 2),  dim3(256), 0, stream, apack, wpack, Bm, scores);
  hipLaunchKernelGGL(finalize_kernel, dim3(2048),    dim3(256), 0, stream,
                     A, Bm, Vw, Vb, Ow, Ob, scores, out);
}
